// AttentionBasedMerger_61083024884008
// MI455X (gfx1250) — matrix operand order repaired and checked by reading, not compiled, not run on hardware
//
#include <hip/hip_runtime.h>
#include <hip/hip_bf16.h>
#include <stdint.h>

// Problem constants (reference: B,SQ,SK,HID,NH = 64,1,2048,1024,16)
#define B_   64
#define SK_  2048
#define HID_ 1024
#define NH_  16
#define HD_  64
#define EPS_ 1e-20f

typedef __attribute__((ext_vector_type(16))) __bf16 v16bf;
typedef __attribute__((ext_vector_type(8)))  __bf16 v8bf;
typedef __attribute__((ext_vector_type(8)))  float  v8f;

union V16 { v16bf v; v8bf h[2]; };
union PK4 { __bf16 b[4]; uint2 u; };

#define WMMA_BF16(A, Bv, C, RA) \
  __builtin_amdgcn_wmma_f32_16x16x32_bf16(false, (A), false, (Bv), (short)0, (C), (RA), false)

// ---------------------------------------------------------------------------
// Kernel A: q = x@Wq + bq ; per-head L2-normalize -> qn (B,NH,HD).
// Also zeroes the per-(b,h) probs-sum accumulator for this launch.
// ---------------------------------------------------------------------------
__global__ __launch_bounds__(256) void qproj_kernel(
    const float* __restrict__ x, const float* __restrict__ Wq,
    const float* __restrict__ bq, float* __restrict__ qn,
    float* __restrict__ Ssum) {
  const int b = blockIdx.x;
  const int tid = threadIdx.x;
  __shared__ float xs[HID_];
  __shared__ float qs[HID_];
  __shared__ float nr[NH_];

  for (int i = tid; i < HID_; i += 256) xs[i] = x[b * HID_ + i];
  if (tid < NH_) Ssum[b * NH_ + tid] = 0.0f;
  __syncthreads();

  float acc[4];
#pragma unroll
  for (int r = 0; r < 4; ++r) acc[r] = bq[tid + r * 256];
  for (int d = 0; d < HID_; ++d) {
    const float xv = xs[d];
    const float* wrow = Wq + (size_t)d * HID_;
#pragma unroll
    for (int r = 0; r < 4; ++r) acc[r] = fmaf(xv, wrow[tid + r * 256], acc[r]);
  }
#pragma unroll
  for (int r = 0; r < 4; ++r) qs[tid + r * 256] = acc[r];
  __syncthreads();

  if (tid < NH_) {
    float ss = 0.f;
#pragma unroll 8
    for (int c = 0; c < HD_; ++c) { float v = qs[tid * HD_ + c]; ss = fmaf(v, v, ss); }
    nr[tid] = rsqrtf(ss);
  }
  __syncthreads();
#pragma unroll
  for (int r = 0; r < 4; ++r) {
    const int i = tid + r * 256;
    qn[(size_t)b * HID_ + i] = qs[i] * nr[i >> 6];
  }
}

// ---------------------------------------------------------------------------
// Kernel B: per (b,h, 128-row tile): K = R_b @ Wk[:,h] via bf16 WMMA,
// add bk, cosine score vs qn, gumbel-softmax gate -> probs(B,NH,SK).
// Register double-buffered global->LDS staging; native bf16 converts;
// batched fragment loads; reuse_a hints on the WMMA group.
// ---------------------------------------------------------------------------
__global__ __launch_bounds__(256) void kscore_kernel(
    const float* __restrict__ R, const float* __restrict__ Wk,
    const float* __restrict__ bk, const float* __restrict__ u_noise,
    const float* __restrict__ qn, float* __restrict__ probs,
    float* __restrict__ Ssum) {
  const int mb  = blockIdx.x * 128;   // row tile base within SK
  const int h   = blockIdx.y;
  const int b   = blockIdx.z;
  const int tid = threadIdx.x;
  const int lane = tid & 31;
  const int wv   = tid >> 5;

  __shared__ __align__(16) __bf16 Asl[128][40];  // R tile bf16, row-major [m][k]
  __shared__ __align__(16) __bf16 Bsl[64][40];   // Wk tile bf16, transposed [n][k]
  __shared__ float qn_s[HD_];
  __shared__ float bk_s[HD_];
  __shared__ float krow[128][65];                // k output, padded (conflict-free)
  __shared__ float psum;

  if (tid < HD_) {
    qn_s[tid] = qn[((size_t)b * NH_ + h) * HD_ + tid];
    bk_s[tid] = bk[h * HD_ + tid];
  }
  if (tid == 0) psum = 0.f;

  const float* Rb = R + (size_t)b * SK_ * HID_;

  float4 ra[4];   // staged R sub-tile (128 x 32 / 256 threads)
  float4 rw[2];   // staged Wk sub-tile (32 x 64 / 256 threads)
  auto stage = [&](int kc) {
#pragma unroll
    for (int i = 0; i < 4; ++i) {
      const int g = tid + 256 * i;
      ra[i] = *reinterpret_cast<const float4*>(
          &Rb[(size_t)(mb + (g >> 3)) * HID_ + kc + (g & 7) * 4]);
    }
#pragma unroll
    for (int i = 0; i < 2; ++i) {
      const int g = tid + 256 * i;
      rw[i] = *reinterpret_cast<const float4*>(
          &Wk[(size_t)(kc + (g >> 4)) * HID_ + h * HD_ + (g & 15) * 4]);
    }
  };

  v8f cacc[4] = {};
  stage(0);

  for (int kc = 0; kc < HID_; kc += 32) {
    // Commit staged chunk to LDS with native bf16 converts
#pragma unroll
    for (int i = 0; i < 4; ++i) {
      const int g = tid + 256 * i;
      const int row = g >> 3;
      const int c4 = (g & 7) * 4;
      PK4 p;
      p.b[0] = (__bf16)ra[i].x; p.b[1] = (__bf16)ra[i].y;
      p.b[2] = (__bf16)ra[i].z; p.b[3] = (__bf16)ra[i].w;
      *reinterpret_cast<uint2*>(&Asl[row][c4]) = p.u;
    }
#pragma unroll
    for (int i = 0; i < 2; ++i) {
      const int g = tid + 256 * i;
      const int kr = g >> 4;
      const int c4 = (g & 15) * 4;
      Bsl[c4 + 0][kr] = (__bf16)rw[i].x;
      Bsl[c4 + 1][kr] = (__bf16)rw[i].y;
      Bsl[c4 + 2][kr] = (__bf16)rw[i].z;
      Bsl[c4 + 3][kr] = (__bf16)rw[i].w;
    }
    __syncthreads();

    // Kick off next chunk's global loads; loadcnt wait lands after the WMMAs
    if (kc + 32 < HID_) stage(kc + 32);

    // A fragment (16-bit A 16x32): lane<16 -> K 0..15; else K 16..31
    const int m   = wv * 16 + (lane & 15);
    const int kbA = (lane < 16) ? 0 : 16;
    V16 a;
    a.h[0] = *reinterpret_cast<const v8bf*>(&Asl[m][kbA]);
    a.h[1] = *reinterpret_cast<const v8bf*>(&Asl[m][kbA + 8]);

    // B fragments (32x16): lane<16 -> K 0..15 ; else K 16..31 ; n = lane%16
    const int kbB = (lane < 16) ? 0 : 16;
    const int n   = lane & 15;
    V16 bb0, bb1, bb2, bb3;
    bb0.h[0] = *reinterpret_cast<const v8bf*>(&Bsl[ 0 + n][kbB]);
    bb0.h[1] = *reinterpret_cast<const v8bf*>(&Bsl[ 0 + n][kbB + 8]);
    bb1.h[0] = *reinterpret_cast<const v8bf*>(&Bsl[16 + n][kbB]);
    bb1.h[1] = *reinterpret_cast<const v8bf*>(&Bsl[16 + n][kbB + 8]);
    bb2.h[0] = *reinterpret_cast<const v8bf*>(&Bsl[32 + n][kbB]);
    bb2.h[1] = *reinterpret_cast<const v8bf*>(&Bsl[32 + n][kbB + 8]);
    bb3.h[0] = *reinterpret_cast<const v8bf*>(&Bsl[48 + n][kbB]);
    bb3.h[1] = *reinterpret_cast<const v8bf*>(&Bsl[48 + n][kbB + 8]);

    cacc[0] = WMMA_BF16(a.v, bb0.v, cacc[0], false);
    cacc[1] = WMMA_BF16(a.v, bb1.v, cacc[1], true);   // same A as previous -> reuse_a
    cacc[2] = WMMA_BF16(a.v, bb2.v, cacc[2], true);
    cacc[3] = WMMA_BF16(a.v, bb3.v, cacc[3], true);
    __syncthreads();
  }

  // Spill k rows to LDS per the C/D layout: VGPR r, lane l -> row r+8*(l/16), col l%16
  {
    const int m0 = wv * 16 + ((lane >> 4) << 3);
    const int col = lane & 15;
#pragma unroll
    for (int r = 0; r < 8; ++r) {
      krow[m0 + r][col +  0] = cacc[0][r];
      krow[m0 + r][col + 16] = cacc[1][r];
      krow[m0 + r][col + 32] = cacc[2][r];
      krow[m0 + r][col + 48] = cacc[3][r];
    }
  }
  __syncthreads();

  // Score + gumbel gate: one thread per key row
  if (tid < 128) {
    const int j = mb + tid;
    float ss = 0.f, dt = 0.f;
#pragma unroll 8
    for (int c = 0; c < HD_; ++c) {
      const float kv = krow[tid][c] + bk_s[c];
      ss = fmaf(kv, kv, ss);
      dt = fmaf(kv, qn_s[c], dt);
    }
    const float score = dt * rsqrtf(ss);         // qn . kn
    const float p = (score + 1.f) * 0.5f;
    const float2 u = *reinterpret_cast<const float2*>(
        &u_noise[(((size_t)b * NH_ + h) * SK_ + j) * 2]);
    const float g0 = -__logf(-__logf(u.x + EPS_) + EPS_);
    const float g1 = -__logf(-__logf(u.y + EPS_) + EPS_);
    const float a0 = __logf(p + EPS_) + g0;
    const float a1 = __logf(1.f - p + EPS_) + g1;
    const float pr = 1.f / (1.f + __expf(a1 - a0));   // softmax([a0,a1])[0], tau=1
    probs[((size_t)b * NH_ + h) * SK_ + j] = pr;
    atomicAdd(&psum, pr);
  }
  __syncthreads();
  if (tid == 0) atomicAdd(&Ssum[b * NH_ + h], psum);
}

// ---------------------------------------------------------------------------
// Kernel C: t_b = probs_b (16 x 2048) @ R_b (2048 x 1024) via bf16 WMMA.
// M=16 == one WMMA tile row; 8 waves x 8 N-tiles = full 1024 N per block.
// ---------------------------------------------------------------------------
__global__ __launch_bounds__(256) void tgemm_kernel(
    const float* __restrict__ R, const float* __restrict__ probs,
    float* __restrict__ tout) {
  const int b   = blockIdx.x;
  const int tid = threadIdx.x;
  const int lane = tid & 31;
  const int wv   = tid >> 5;

  __shared__ __align__(16) __bf16 Apl[16][40];     // probs chunk (16 x 32)
  __shared__ __align__(16) __bf16 Rpl[1024][40];   // R chunk transposed [n][k]

  const float* Rb = R + (size_t)b * SK_ * HID_;
  const float* Pb = probs + (size_t)b * NH_ * SK_;
  v8f acc[8] = {};

  for (int jc = 0; jc < SK_; jc += 32) {
    __syncthreads();
    {
      const int g  = tid * 2;                // 512 values = 16h x 32j
      const int hh = g >> 5;
      const int jj = g & 31;
      PK4 p;
      p.b[0] = (__bf16)Pb[(size_t)hh * SK_ + jc + jj];
      p.b[1] = (__bf16)Pb[(size_t)hh * SK_ + jc + jj + 1];
      *reinterpret_cast<uint32_t*>(&Apl[hh][jj]) = p.u.x;
    }
    // Prefetch a slice of the next K-chunk (global_prefetch_b8)
    if (jc + 32 < SK_) {
      __builtin_prefetch(&Rb[(size_t)(jc + 32 + (tid >> 3)) * HID_ + (tid & 7) * 128], 0, 1);
    }
#pragma unroll 4
    for (int i = 0; i < 32; ++i) {           // 32 K-rows x 1024 N-cols, transposed store
      const float4 f = *reinterpret_cast<const float4*>(
          &Rb[(size_t)(jc + i) * HID_ + tid * 4]);
      const int n = tid * 4;
      Rpl[n + 0][i] = (__bf16)f.x;
      Rpl[n + 1][i] = (__bf16)f.y;
      Rpl[n + 2][i] = (__bf16)f.z;
      Rpl[n + 3][i] = (__bf16)f.w;
    }
    __syncthreads();

    const int kbA = (lane < 16) ? 0 : 16;
    const int hh  = lane & 15;
    V16 a;
    a.h[0] = *reinterpret_cast<const v8bf*>(&Apl[hh][kbA]);
    a.h[1] = *reinterpret_cast<const v8bf*>(&Apl[hh][kbA + 8]);
    const int kbB = (lane < 16) ? 0 : 16;
    const int n0  = wv * 128 + (lane & 15);

    V16 bb[8];
#pragma unroll
    for (int t = 0; t < 8; ++t) {
      bb[t].h[0] = *reinterpret_cast<const v8bf*>(&Rpl[n0 + t * 16][kbB]);
      bb[t].h[1] = *reinterpret_cast<const v8bf*>(&Rpl[n0 + t * 16][kbB + 8]);
    }
    acc[0] = WMMA_BF16(a.v, bb[0].v, acc[0], false);
    acc[1] = WMMA_BF16(a.v, bb[1].v, acc[1], true);
    acc[2] = WMMA_BF16(a.v, bb[2].v, acc[2], true);
    acc[3] = WMMA_BF16(a.v, bb[3].v, acc[3], true);
    acc[4] = WMMA_BF16(a.v, bb[4].v, acc[4], true);
    acc[5] = WMMA_BF16(a.v, bb[5].v, acc[5], true);
    acc[6] = WMMA_BF16(a.v, bb[6].v, acc[6], true);
    acc[7] = WMMA_BF16(a.v, bb[7].v, acc[7], true);
  }

  float* tb = tout + (size_t)b * NH_ * HID_;
  const int rowoff = (lane >> 4) << 3;
#pragma unroll
  for (int t = 0; t < 8; ++t) {
    const int d = wv * 128 + t * 16 + (lane & 15);
#pragma unroll
    for (int r = 0; r < 8; ++r) {
      tb[(size_t)(r + rowoff) * HID_ + d] = acc[t][r];
    }
  }
}

// ---------------------------------------------------------------------------
// Kernel D: ctx = t @ Wv + S*bv  (per-head columns), out = ctx @ Wd + bd.
// ---------------------------------------------------------------------------
__global__ __launch_bounds__(256) void out_kernel(
    const float* __restrict__ tin, const float* __restrict__ Ssum,
    const float* __restrict__ Wv, const float* __restrict__ bv,
    const float* __restrict__ Wd, const float* __restrict__ bd,
    float* __restrict__ out) {
  const int b = blockIdx.x;
  const int tid = threadIdx.x;
  __shared__ float ts[NH_ * HID_];   // 64 KB
  __shared__ float ctxs[HID_];
  __shared__ float Ss[NH_];

  for (int i = tid; i < NH_ * HID_; i += 256) ts[i] = tin[(size_t)b * NH_ * HID_ + i];
  if (tid < NH_) Ss[tid] = Ssum[b * NH_ + tid];
  __syncthreads();

  float acc[4];
#pragma unroll
  for (int r = 0; r < 4; ++r) {
    const int i = tid + r * 256;
    acc[r] = Ss[i >> 6] * bv[i];     // (sum_j probs_j) * bv, column i belongs to head i/64
  }
  for (int d = 0; d < HID_; ++d) {
    const float* wrow = Wv + (size_t)d * HID_;
#pragma unroll
    for (int r = 0; r < 4; ++r) {
      const int i = tid + r * 256;
      acc[r] = fmaf(ts[(i >> 6) * HID_ + d], wrow[i], acc[r]);
    }
  }
#pragma unroll
  for (int r = 0; r < 4; ++r) ctxs[tid + r * 256] = acc[r];
  __syncthreads();

  float o[4];
#pragma unroll
  for (int r = 0; r < 4; ++r) o[r] = bd[tid + r * 256];
  for (int i = 0; i < HID_; ++i) {
    const float cv = ctxs[i];
    const float* wrow = Wd + (size_t)i * HID_;
#pragma unroll
    for (int r = 0; r < 4; ++r) o[r] = fmaf(cv, wrow[tid + r * 256], o[r]);
  }
#pragma unroll
  for (int r = 0; r < 4; ++r) out[(size_t)b * HID_ + tid + r * 256] = o[r];
}

// ---------------------------------------------------------------------------
extern "C" void kernel_launch(void* const* d_in, const int* in_sizes, int n_in,
                              void* d_out, int out_size, void* d_ws, size_t ws_size,
                              hipStream_t stream) {
  (void)in_sizes; (void)n_in; (void)out_size; (void)ws_size;
  const float* x  = (const float*)d_in[0];   // input_tensor  (B,1,HID)
  const float* R  = (const float*)d_in[1];   // retrieval     (B,SK,HID)
  const float* un = (const float*)d_in[2];   // u_noise       (B,NH,1,SK,2)
  const float* Wq = (const float*)d_in[3];
  const float* bq = (const float*)d_in[4];
  const float* Wk = (const float*)d_in[5];
  const float* bk = (const float*)d_in[6];
  const float* Wv = (const float*)d_in[7];
  const float* bv = (const float*)d_in[8];
  const float* Wd = (const float*)d_in[9];
  const float* bd = (const float*)d_in[10];
  float* out = (float*)d_out;

  char* ws = (char*)d_ws;
  float* qn    = (float*)(ws);                                        // B*HID
  float* Ssum  = (float*)(ws + (size_t)B_ * HID_ * 4);                // B*NH
  float* probs = (float*)(ws + (size_t)B_ * HID_ * 4 + 4096);         // B*NH*SK
  float* tbuf  = (float*)((char*)probs + (size_t)B_ * NH_ * SK_ * 4); // B*NH*HID

  qproj_kernel<<<B_, 256, 0, stream>>>(x, Wq, bq, qn, Ssum);
  kscore_kernel<<<dim3(SK_ / 128, NH_, B_), 256, 0, stream>>>(
      R, Wk, bk, un, qn, probs, Ssum);
  tgemm_kernel<<<B_, 256, 0, stream>>>(R, probs, tbuf);
  out_kernel<<<B_, 256, 0, stream>>>(tbuf, Ssum, Wv, bv, Wd, bd, out);
}
